// Classifier_13417477832960
// MI455X (gfx1250) — matrix operand order repaired and checked by reading, not compiled, not run on hardware
//
#include <hip/hip_runtime.h>
#include <hip/hip_bf16.h>
#include <hip/hip_fp16.h>

typedef __attribute__((ext_vector_type(16))) _Float16 v16h;
typedef __attribute__((ext_vector_type(8)))  float    v8f;

#define N_NODES   100000
#define N_EDGES   1600000
#define N_GRAPHS  128
#define HIDDEN    128
#define N_CLASSES 10

// ---------------- utility: zero a float region ----------------
__global__ void k_zero(float* __restrict__ p, int n) {
    int i = blockIdx.x * blockDim.x + threadIdx.x;
    if (i < n) p[i] = 0.0f;
}

// ---------------- degree computation (fp32 atomics) ----------------
__global__ void k_deg(const int* __restrict__ src, const int* __restrict__ dst,
                      float* __restrict__ indeg, float* __restrict__ outdeg) {
    int e = blockIdx.x * blockDim.x + threadIdx.x;
    if (e < N_EDGES) {
        atomicAdd(&outdeg[src[e]], 1.0f);
        atomicAdd(&indeg[dst[e]], 1.0f);
    }
}

// ---------------- norms + layer0 feature + graph counts ----------------
__global__ void k_norm(const int* __restrict__ gid,
                       const float* __restrict__ indeg, const float* __restrict__ outdeg,
                       float* __restrict__ ns, float* __restrict__ nd,
                       float* __restrict__ h0s, float* __restrict__ counts) {
    int n = blockIdx.x * blockDim.x + threadIdx.x;
    if (n < N_NODES) {
        float od = outdeg[n], idg = indeg[n];
        float s = 1.0f / sqrtf(fmaxf(od, 1.0f));
        float d = 1.0f / sqrtf(fmaxf(idg, 1.0f));
        ns[n] = s;
        nd[n] = d;
        h0s[n] = idg * s;                 // h0 * norm_src, ready for gather
        atomicAdd(&counts[gid[n]], 1.0f);
    }
}

// ---------------- W2 -> f16 transposed (B-fragment friendly) ----------------
__global__ void k_w2cvt(const float* __restrict__ W2, _Float16* __restrict__ W2t) {
    int i = blockIdx.x * blockDim.x + threadIdx.x;   // 16384
    if (i < HIDDEN * HIDDEN) {
        int k = i / HIDDEN, c = i % HIDDEN;
        W2t[c * HIDDEN + k] = (_Float16)W2[i];       // W2t[col][k]
    }
}

// ---------------- layer-1 scalar scatter ----------------
__global__ void k_scat1(const int* __restrict__ src, const int* __restrict__ dst,
                        const float* __restrict__ h0s, float* __restrict__ agg1) {
    int e = blockIdx.x * blockDim.x + threadIdx.x;
    if (e < N_EDGES) atomicAdd(&agg1[dst[e]], h0s[src[e]]);
}

// ---------------- layer-1 node update -> h1s (f16, pre-scaled by norm_src) ----
// also zero-initializes agg2 (f16)
__global__ void k_node1(const float* __restrict__ agg1,
                        const float* __restrict__ ns, const float* __restrict__ nd,
                        const float* __restrict__ W1, const float* __restrict__ b1,
                        _Float16* __restrict__ h1s, _Float16* __restrict__ agg2) {
    int t = blockIdx.x * 256 + threadIdx.x;          // [0, N_NODES*128)
    int n = t >> 7;
    int j = t & 127;
    float a = agg1[n] * nd[n];
    float v = fmaf(a, W1[j], b1[j]);
    v = fmaxf(v, 0.0f) * ns[n];                      // relu, pre-scale for next gather
    h1s[(size_t)n * HIDDEN + j] = (_Float16)v;
    agg2[(size_t)n * HIDDEN + j] = (_Float16)0.0f;
}

// ---------------- packed f16 atomic add (ISA: GLOBAL_ATOMIC_PK_ADD_F16) ------
__device__ __forceinline__ void pk_add_f16(unsigned int* addr, unsigned int val) {
    asm volatile("global_atomic_pk_add_f16 %0, %1, off"
                 :: "v"(addr), "v"(val) : "memory");
}

// ---------------- layer-2 feature scatter: one wave per edge -----------------
__global__ void k_scat2(const int* __restrict__ src, const int* __restrict__ dst,
                        const unsigned int* __restrict__ h1s32,
                        unsigned int* __restrict__ agg2_32) {
    int w = (blockIdx.x << 3) + (threadIdx.x >> 5);  // edge id (8 waves/block)
    int lane = threadIdx.x & 31;
    if (w < N_EDGES) {
        int s = src[w], d = dst[w];
        const unsigned int* hr = h1s32 + (size_t)s * 64;   // 128 f16 = 64 dwords
        unsigned int* ar = agg2_32 + (size_t)d * 64;
        unsigned int v0 = hr[lane];
        unsigned int v1 = hr[lane + 32];
        if (v0) pk_add_f16(ar + lane, v0);                 // skip all-zero pairs
        if (v1) pk_add_f16(ar + lane + 32, v1);
    }
}

// ---------------- layer-2 GEMM via WMMA + fused graph pooling ----------------
// one wave per 16-node tile; 8 N-tiles x 4 K-steps = 32 wmma per wave
__global__ void k_gemm2(const _Float16* __restrict__ agg2,
                        const _Float16* __restrict__ W2t,
                        const float* __restrict__ nd, const float* __restrict__ b2,
                        const int* __restrict__ gid, float* __restrict__ gsum) {
    int wave = (blockIdx.x << 3) + (threadIdx.x >> 5);
    int tb = wave << 4;                              // first node of tile
    if (tb >= N_NODES) return;                       // wave-uniform exit, EXEC stays all-1
    int lane = threadIdx.x & 31;
    int nl = lane & 15;
    int kh = lane >> 4;

    union V { v16h h; uint4 u[2]; };
    v8f acc[8] = {};

    const _Float16* arow = agg2 + (size_t)(tb + nl) * HIDDEN;
#pragma unroll
    for (int k = 0; k < 4; k++) {                    // K = 128 in steps of 32
        V A;
        // 16-bit A layout: lane<16 row M=nl holds K {0..7} then {16..23}; lane>=16 K+8
        A.u[0] = *reinterpret_cast<const uint4*>(arow + k * 32 + kh * 8);
        A.u[1] = *reinterpret_cast<const uint4*>(arow + k * 32 + 16 + kh * 8);
#pragma unroll
        for (int nt = 0; nt < 8; nt++) {
            V B;
            // 16-bit B layout: lane -> column nt*16+nl; K {0..7} then {16..23}, lane>=16 K+8
            const _Float16* brow = W2t + (size_t)(nt * 16 + nl) * HIDDEN + k * 32 + kh * 8;
            B.u[0] = *reinterpret_cast<const uint4*>(brow);
            B.u[1] = *reinterpret_cast<const uint4*>(brow + 16);
            acc[nt] = __builtin_amdgcn_wmma_f32_16x16x32_f16(
                false, A.h, false, B.h, (short)0, acc[nt], false, false);
        }
    }

    float b2v[8];
#pragma unroll
    for (int nt = 0; nt < 8; nt++) b2v[nt] = b2[nt * 16 + nl];

    // D layout: VGPR r -> node tb + r + 8*kh; lane -> feature col nt*16+nl
#pragma unroll
    for (int r = 0; r < 8; r++) {
        int node = tb + r + (kh << 3);
        float d = nd[node];
        float* gs = gsum + (size_t)gid[node] * HIDDEN;
#pragma unroll
        for (int nt = 0; nt < 8; nt++) {
            float v = fmaf(acc[nt][r], d, b2v[nt]);
            if (v > 0.0f) atomicAdd(gs + nt * 16 + nl, v);   // relu + pooled sum
        }
    }
}

// ---------------- classifier: out = (gsum/cnt) @ Wc + bc ----------------
__global__ void k_out(const float* __restrict__ gsum, const float* __restrict__ counts,
                      const float* __restrict__ Wc, const float* __restrict__ bc,
                      float* __restrict__ out) {
    int t = blockIdx.x * blockDim.x + threadIdx.x;
    if (t < N_GRAPHS * N_CLASSES) {
        int g = t / N_CLASSES, c = t % N_CLASSES;
        float inv = 1.0f / fmaxf(counts[g], 1.0f);
        const float* row = gsum + (size_t)g * HIDDEN;
        float s = 0.0f;
#pragma unroll 8
        for (int k = 0; k < HIDDEN; k++) s = fmaf(row[k], Wc[k * N_CLASSES + c], s);
        out[t] = fmaf(s, inv, bc[c]);
    }
}

extern "C" void kernel_launch(void* const* d_in, const int* in_sizes, int n_in,
                              void* d_out, int out_size, void* d_ws, size_t ws_size,
                              hipStream_t stream) {
    const int*   src = (const int*)d_in[0];
    const int*   dst = (const int*)d_in[1];
    const int*   gid = (const int*)d_in[2];
    const float* W1  = (const float*)d_in[3];
    const float* b1  = (const float*)d_in[4];
    const float* W2  = (const float*)d_in[5];
    const float* b2  = (const float*)d_in[6];
    const float* Wc  = (const float*)d_in[7];
    const float* bc  = (const float*)d_in[8];

    char* ws = (char*)d_ws;
    // layout (bytes), all offsets 16B aligned
    float*    indeg  = (float*)(ws + 0);          //  400000
    float*    outdeg = (float*)(ws + 400000);     //  400000
    float*    agg1   = (float*)(ws + 800000);     //  400000
    float*    gsum   = (float*)(ws + 1200000);    //   65536
    float*    counts = (float*)(ws + 1265536);    //     512
    float*    ns     = (float*)(ws + 1266048);    //  400000
    float*    nd     = (float*)(ws + 1666048);    //  400000
    float*    h0s    = (float*)(ws + 2066048);    //  400000
    _Float16* h1s    = (_Float16*)(ws + 2466048); // 25600000
    _Float16* agg2   = (_Float16*)(ws + 28066048);// 25600000
    _Float16* W2t    = (_Float16*)(ws + 53666048);//   32768

    const int zeroN = 1266048 / 4;  // indeg+outdeg+agg1+gsum+counts

    k_zero <<<(zeroN + 255) / 256, 256, 0, stream>>>((float*)ws, zeroN);
    k_deg  <<<(N_EDGES + 255) / 256, 256, 0, stream>>>(src, dst, indeg, outdeg);
    k_norm <<<(N_NODES + 255) / 256, 256, 0, stream>>>(gid, indeg, outdeg, ns, nd, h0s, counts);
    k_w2cvt<<<64, 256, 0, stream>>>(W2, W2t);
    k_scat1<<<(N_EDGES + 255) / 256, 256, 0, stream>>>(src, dst, h0s, agg1);
    k_node1<<<(N_NODES * HIDDEN) / 256, 256, 0, stream>>>(agg1, ns, nd, W1, b1, h1s, agg2);
    k_scat2<<<N_EDGES / 8, 256, 0, stream>>>(src, dst,
                                             (const unsigned int*)h1s, (unsigned int*)agg2);
    k_gemm2<<<(N_NODES / 16 + 7) / 8, 256, 0, stream>>>(agg2, W2t, nd, b2, gid, gsum);
    k_out  <<<5, 256, 0, stream>>>(gsum, counts, Wc, bc, (float*)d_out);
}
